// GraphAllocNet_37761352467077
// MI455X (gfx1250) — compile-verified, matrix operand order checked by reading, not run on hardware
//
#include <hip/hip_runtime.h>
#include <hip/hip_bf16.h>
#include <math.h>

// ---------------------------------------------------------------------------
// GraphAllocNet on MI455X (gfx1250): all dense layers as wave32 WMMA
// f32_16x16x32_f16 GEMMs with f32 accumulation; gathers fused into A-fragment
// loads; segment_sum as hardware f32 atomics; readout GEMV fused via LDS.
// ---------------------------------------------------------------------------

#define HID 128
#define N_NODES 20000
#define N_EDGES 640000
#define N_DEMANDS 100000
#define NODE_TILES (N_NODES / 16)     // 1250
#define EDGE_TILES (N_EDGES / 16)     // 40000
#define DEM_TILES  (N_DEMANDS / 16)   // 6250

typedef __attribute__((ext_vector_type(16))) _Float16 v16h;
typedef __attribute__((ext_vector_type(8)))  _Float16 v8h;
typedef __attribute__((ext_vector_type(8)))  float    v8f;

// --- WMMA wrapper: D = A(16x32 f16) * B(32x16 f16) + C(16x16 f32) ----------
__device__ __forceinline__ v8f wmma_f16(v16h a, v16h b, v8f c) {
  return __builtin_amdgcn_wmma_f32_16x16x32_f16(
      /*neg_a=*/false, a, /*neg_b=*/false, b,
      /*c_mod=*/(short)0, c, /*reuse_a=*/false, /*reuse_b=*/false);
}

// --- A fragment: 16x32 f16 tile, this lane = row (lane&15), halves by lane>>4.
// ISA layout: lanes 0-15 hold K = [h*8 .. h*8+7] (v0-3) and [16+h*8 ..] (v4-7).
// Two 16B contiguous loads per lane.
__device__ __forceinline__ v16h ld_a16(const _Float16* __restrict__ row,
                                       int k0, int half) {
  union { v16h v; v8h h[2]; } u;
  u.h[0] = *(const v8h*)(row + k0 + half * 8);
  u.h[1] = *(const v8h*)(row + k0 + 16 + half * 8);
  return u.v;
}

// --- A fragment from an f32 row (convert on the fly) -----------------------
__device__ __forceinline__ v16h ld_a32(const float* __restrict__ row,
                                       int k0, int half) {
  v16h a;
  const float* p0 = row + k0 + half * 8;
  const float* p1 = row + k0 + 16 + half * 8;
#pragma unroll
  for (int j = 0; j < 8; ++j) {
    a[j]     = (_Float16)p0[j];
    a[8 + j] = (_Float16)p1[j];
  }
  return a;
}

// --- B fragment: 32x16 slice of row-major W[K x 128] f16; this lane = col n.
__device__ __forceinline__ v16h ld_b(const _Float16* __restrict__ W,
                                     int k0, int n, int half) {
  v16h b;
  const _Float16* p = W + (size_t)(k0 + half * 8) * HID + n;
#pragma unroll
  for (int j = 0; j < 8; ++j) {
    b[j]     = p[(size_t)j * HID];
    b[8 + j] = p[(size_t)(16 + j) * HID];
  }
  return b;
}

// ---------------------------------------------------------------------------
// Weight convert f32 -> f16 with zero padding (row-major, pad = tail elems)
// ---------------------------------------------------------------------------
__global__ void cvt_w_kernel(const float* __restrict__ src,
                             _Float16* __restrict__ dst,
                             int n_src, int n_dst) {
  int i = blockIdx.x * blockDim.x + threadIdx.x;
  if (i < n_dst) dst[i] = (i < n_src) ? (_Float16)src[i] : (_Float16)0.f;
}

// ---------------------------------------------------------------------------
// Encoder: H[M x 128] = relu(X[M x KSRC] @ W16[32(padded) x 128] + bias)
// KSRC in {32, 16}; one WMMA ktile, zeros beyond KSRC.
// ---------------------------------------------------------------------------
template <int KSRC>
__global__ void encode_kernel(const float* __restrict__ X,
                              const _Float16* __restrict__ W16,
                              const float* __restrict__ bias,
                              _Float16* __restrict__ H, int m_tiles) {
  const int lane = threadIdx.x & 31;
  const int wave = threadIdx.x >> 5;
  const int half = lane >> 4;
  const int nl   = lane & 15;
  const int n    = wave * 16 + nl;

  const v16h b = ld_b(W16, 0, n, half);
  const float bs = bias[n];

  for (int t = blockIdx.x; t < m_tiles; t += gridDim.x) {
    const float* xr = X + (size_t)(t * 16 + nl) * KSRC;
    v16h a;
#pragma unroll
    for (int j = 0; j < 8; ++j) {
      int k  = half * 8 + j;
      int k2 = 16 + half * 8 + j;
      a[j]     = (k  < KSRC) ? (_Float16)xr[k]  : (_Float16)0.f;
      a[8 + j] = (k2 < KSRC) ? (_Float16)xr[k2] : (_Float16)0.f;
    }
    v8f c = {};
    c = wmma_f16(a, b, c);
#pragma unroll
    for (int r = 0; r < 8; ++r) {
      float v = c[r] + bs;
      v = v > 0.f ? v : 0.f;
      H[(size_t)(t * 16 + r + half * 8) * HID + n] = (_Float16)v;
    }
  }
}

// ---------------------------------------------------------------------------
// Message pass: msg = relu(h[src]·Wm[0:128] + hE·Wm[128:256] + h[dst]·Wm[256:384] + b)
// then agg[dst] += msg (f32 HW atomics). Messages never materialized.
// ---------------------------------------------------------------------------
__global__ void msg_kernel(const _Float16* __restrict__ hN,
                           const _Float16* __restrict__ hE,
                           const int* __restrict__ src,
                           const int* __restrict__ dst,
                           const _Float16* __restrict__ Wm,  // 384 x 128 f16
                           const float* __restrict__ bias,
                           float* __restrict__ agg) {
  const int lane = threadIdx.x & 31;
  const int wave = threadIdx.x >> 5;
  const int half = lane >> 4;
  const int nl   = lane & 15;
  const int n    = wave * 16 + nl;

  v16h B[12];
#pragma unroll
  for (int k = 0; k < 12; ++k) B[k] = ld_b(Wm, k * 32, n, half);
  const float bs = bias[n];

  for (int t = blockIdx.x; t < EDGE_TILES; t += gridDim.x) {
    const int e = t * 16 + nl;
    const _Float16* rs = hN + (size_t)src[e] * HID;
    const _Float16* re = hE + (size_t)e * HID;
    const _Float16* rd = hN + (size_t)dst[e] * HID;

    v8f c = {};
#pragma unroll
    for (int k = 0; k < 4; ++k) c = wmma_f16(ld_a16(rs, k * 32, half), B[k], c);
#pragma unroll
    for (int k = 0; k < 4; ++k) c = wmma_f16(ld_a16(re, k * 32, half), B[4 + k], c);
#pragma unroll
    for (int k = 0; k < 4; ++k) c = wmma_f16(ld_a16(rd, k * 32, half), B[8 + k], c);

#pragma unroll
    for (int r = 0; r < 8; ++r) {
      float v = c[r] + bs;
      v = v > 0.f ? v : 0.f;
      const int e2 = t * 16 + r + half * 8;
      unsafeAtomicAdd(&agg[(size_t)dst[e2] * HID + n], v);
    }
  }
}

// ---------------------------------------------------------------------------
// Node update: hOut = relu(h·Wu[0:128] + agg·Wu[128:256] + b)
// ---------------------------------------------------------------------------
__global__ void upd_kernel(const _Float16* __restrict__ hN,
                           const float* __restrict__ agg,
                           const _Float16* __restrict__ Wu,  // 256 x 128 f16
                           const float* __restrict__ bias,
                           _Float16* __restrict__ hOut) {
  const int lane = threadIdx.x & 31;
  const int wave = threadIdx.x >> 5;
  const int half = lane >> 4;
  const int nl   = lane & 15;
  const int n    = wave * 16 + nl;

  v16h B[8];
#pragma unroll
  for (int k = 0; k < 8; ++k) B[k] = ld_b(Wu, k * 32, n, half);
  const float bs = bias[n];

  for (int t = blockIdx.x; t < NODE_TILES; t += gridDim.x) {
    const int row = t * 16 + nl;
    const _Float16* rh = hN + (size_t)row * HID;
    const float*    ra = agg + (size_t)row * HID;

    v8f c = {};
#pragma unroll
    for (int k = 0; k < 4; ++k) c = wmma_f16(ld_a16(rh, k * 32, half), B[k], c);
#pragma unroll
    for (int k = 0; k < 4; ++k) c = wmma_f16(ld_a32(ra, k * 32, half), B[4 + k], c);

#pragma unroll
    for (int r = 0; r < 8; ++r) {
      float v = c[r] + bs;
      v = v > 0.f ? v : 0.f;
      hOut[(size_t)(t * 16 + r + half * 8) * HID + n] = (_Float16)v;
    }
  }
}

// ---------------------------------------------------------------------------
// Readout: hid = relu([h[dsrc]|h[ddst]|dem] @ Wr1 + b1), out = sigmoid(hid·Wr2 + b2)
// Wr1 padded to 288 rows (264 valid). Final GEMV fused via LDS reduction.
// ---------------------------------------------------------------------------
__global__ void readout_kernel(const _Float16* __restrict__ hN,
                               const float* __restrict__ dem,   // [D x 8]
                               const int* __restrict__ pairs,   // [D x 2]
                               const _Float16* __restrict__ Wr1, // 288 x 128 f16
                               const float* __restrict__ br1,
                               const float* __restrict__ Wr2,   // 128
                               const float* __restrict__ br2,   // 1
                               float* __restrict__ out) {
  __shared__ float acc[16];
  const int lane = threadIdx.x & 31;
  const int wave = threadIdx.x >> 5;
  const int half = lane >> 4;
  const int nl   = lane & 15;
  const int n    = wave * 16 + nl;

  v16h B[9];
#pragma unroll
  for (int k = 0; k < 9; ++k) B[k] = ld_b(Wr1, k * 32, n, half);
  const float bs = br1[n];
  const float w2 = Wr2[n];
  const float b2 = br2[0];

  for (int t = blockIdx.x; t < DEM_TILES; t += gridDim.x) {
    const int d = t * 16 + nl;
    const _Float16* rs = hN + (size_t)pairs[2 * d + 0] * HID;
    const _Float16* rd = hN + (size_t)pairs[2 * d + 1] * HID;
    const float*    rf = dem + (size_t)d * 8;

    v8f c = {};
#pragma unroll
    for (int k = 0; k < 4; ++k) c = wmma_f16(ld_a16(rs, k * 32, half), B[k], c);
#pragma unroll
    for (int k = 0; k < 4; ++k) c = wmma_f16(ld_a16(rd, k * 32, half), B[4 + k], c);
    // demand tail: K = 8 valid (lanes with half==1 cover K 8..15 -> zero)
    v16h a;
#pragma unroll
    for (int j = 0; j < 8; ++j) {
      a[j]     = (half == 0) ? (_Float16)rf[j] : (_Float16)0.f;
      a[8 + j] = (_Float16)0.f;
    }
    c = wmma_f16(a, B[8], c);

    if (threadIdx.x < 16) acc[threadIdx.x] = 0.f;
    __syncthreads();
#pragma unroll
    for (int r = 0; r < 8; ++r) {
      float v = c[r] + bs;
      v = v > 0.f ? v : 0.f;
      atomicAdd(&acc[r + half * 8], v * w2);   // ds_add_f32
    }
    __syncthreads();
    if (threadIdx.x < 16) {
      float s = acc[threadIdx.x] + b2;
      out[t * 16 + threadIdx.x] = 1.f / (1.f + __expf(-s));
    }
    __syncthreads();
  }
}

// ---------------------------------------------------------------------------
extern "C" void kernel_launch(void* const* d_in, const int* in_sizes, int n_in,
                              void* d_out, int out_size, void* d_ws, size_t ws_size,
                              hipStream_t stream) {
  // Inputs (setup_inputs order)
  const float* node_feats = (const float*)d_in[0];
  const float* edge_feats = (const float*)d_in[1];
  const float* dem_feats  = (const float*)d_in[2];
  const int*   edge_index = (const int*)d_in[3];
  const int*   dem_pairs  = (const int*)d_in[4];
  const float* W_node = (const float*)d_in[5];  const float* b_node = (const float*)d_in[6];
  const float* W_edge = (const float*)d_in[7];  const float* b_edge = (const float*)d_in[8];
  const float* W_msg0 = (const float*)d_in[9];  const float* b_msg0 = (const float*)d_in[10];
  const float* W_upd0 = (const float*)d_in[11]; const float* b_upd0 = (const float*)d_in[12];
  const float* W_msg1 = (const float*)d_in[13]; const float* b_msg1 = (const float*)d_in[14];
  const float* W_upd1 = (const float*)d_in[15]; const float* b_upd1 = (const float*)d_in[16];
  const float* W_r1   = (const float*)d_in[17]; const float* b_r1   = (const float*)d_in[18];
  const float* W_r2   = (const float*)d_in[19]; const float* b_r2   = (const float*)d_in[20];

  const int* src = edge_index;             // edge_index[0]
  const int* dst = edge_index + N_EDGES;   // edge_index[1]
  float* out = (float*)d_out;

  // Workspace carve-up
  char* ws = (char*)d_ws;
  size_t off = 0;
  auto take = [&](size_t bytes) -> char* {
    char* p = ws + off;
    off = (off + bytes + 255) & ~(size_t)255;
    return p;
  };
  _Float16* Wn16  = (_Float16*)take((size_t)32  * HID * 2);
  _Float16* We16  = (_Float16*)take((size_t)32  * HID * 2);   // padded from 16
  _Float16* Wm0h  = (_Float16*)take((size_t)384 * HID * 2);
  _Float16* Wu0h  = (_Float16*)take((size_t)256 * HID * 2);
  _Float16* Wm1h  = (_Float16*)take((size_t)384 * HID * 2);
  _Float16* Wu1h  = (_Float16*)take((size_t)256 * HID * 2);
  _Float16* Wr1h  = (_Float16*)take((size_t)288 * HID * 2);   // padded from 264
  _Float16* h0    = (_Float16*)take((size_t)N_NODES * HID * 2);
  _Float16* h1    = (_Float16*)take((size_t)N_NODES * HID * 2);
  _Float16* hE    = (_Float16*)take((size_t)N_EDGES * HID * 2);
  float*    agg   = (float*)   take((size_t)N_NODES * HID * 4);

  // 1) weights -> f16 (zero-padded)
  auto cvt = [&](const float* s, _Float16* dptr, int ns, int nd) {
    cvt_w_kernel<<<(nd + 255) / 256, 256, 0, stream>>>(s, dptr, ns, nd);
  };
  cvt(W_node, Wn16, 32 * HID, 32 * HID);
  cvt(W_edge, We16, 16 * HID, 32 * HID);
  cvt(W_msg0, Wm0h, 384 * HID, 384 * HID);
  cvt(W_upd0, Wu0h, 256 * HID, 256 * HID);
  cvt(W_msg1, Wm1h, 384 * HID, 384 * HID);
  cvt(W_upd1, Wu1h, 256 * HID, 256 * HID);
  cvt(W_r1,   Wr1h, 264 * HID, 288 * HID);

  // 2) encoders
  encode_kernel<32><<<NODE_TILES, 256, 0, stream>>>(node_feats, Wn16, b_node, h0, NODE_TILES);
  encode_kernel<16><<<2048, 256, 0, stream>>>(edge_feats, We16, b_edge, hE, EDGE_TILES);

  // 3) layer 0: h0 -> h1
  hipMemsetAsync(agg, 0, (size_t)N_NODES * HID * 4, stream);
  msg_kernel<<<2048, 256, 0, stream>>>(h0, hE, src, dst, Wm0h, b_msg0, agg);
  upd_kernel<<<NODE_TILES, 256, 0, stream>>>(h0, agg, Wu0h, b_upd0, h1);

  // 4) layer 1: h1 -> h0
  hipMemsetAsync(agg, 0, (size_t)N_NODES * HID * 4, stream);
  msg_kernel<<<2048, 256, 0, stream>>>(h1, hE, src, dst, Wm1h, b_msg1, agg);
  upd_kernel<<<NODE_TILES, 256, 0, stream>>>(h1, agg, Wu1h, b_upd1, h0);

  // 5) readout (r1 + fused r2 + sigmoid)
  readout_kernel<<<2048, 256, 0, stream>>>(h0, dem_feats, dem_pairs, Wr1h, b_r1,
                                           W_r2, b_r2, out);
}
